// EncCLSTMBlock_32461362823759
// MI455X (gfx1250) — compile-verified, matrix operand order checked by reading, not run on hardware
//
#include <hip/hip_runtime.h>
#include <math.h>

// ---------------------------------------------------------------------------
// CDNA5 (gfx1250) wave32 WMMA implicit-GEMM ConvLSTM stack, round 6.
// - v_wmma_f32_16x16x32_bf16 core, bf16 operand tensors, halo padding.
// - Weight panels pre-permuted once per layer into global staging; hot
//   kernels stage them to LDS with GLOBAL_LOAD_ASYNC_TO_LDS_B128 (+ s_wait_
//   asynccnt), falling back to vector LDS stores if the builtin is absent.
// ---------------------------------------------------------------------------

typedef __bf16 v16bf  __attribute__((ext_vector_type(16)));
typedef __bf16 bf16x8 __attribute__((ext_vector_type(8)));
typedef float  v8f    __attribute__((ext_vector_type(8)));
typedef int    v4i    __attribute__((vector_size(16)));   // matches builtin param

#define AS1 __attribute__((address_space(1)))
#define AS3 __attribute__((address_space(3)))

__device__ __forceinline__ float hsig(float x) {
    return fminf(fmaxf(0.2f * x + 0.5f, 0.0f), 1.0f);
}

__device__ __forceinline__ float fast_tanh(float x) {
#if __has_builtin(__builtin_amdgcn_tanhf)
    return __builtin_amdgcn_tanhf(x);   // v_tanh_f32 (gfx1250 TRANS op)
#else
    return tanhf(x);
#endif
}

__device__ __forceinline__ void wait_async0() {
#if __has_builtin(__builtin_amdgcn_s_wait_asynccnt)
    __builtin_amdgcn_s_wait_asynccnt(0);
#else
    asm volatile("s_wait_asynccnt 0" ::: "memory");
#endif
}

// Stage BYTES of pre-permuted bf16 weight panel from global to LDS.
template <int BYTES>
__device__ __forceinline__ void stage_lds(const __bf16* __restrict__ g,
                                          __bf16* l) {
    constexpr int UNITS = BYTES / 16;          // 16-byte units
#if __has_builtin(__builtin_amdgcn_global_load_async_to_lds_b128)
    AS1 v4i* gp = (AS1 v4i*)(unsigned long long)g;
    AS3 v4i* lp = (AS3 v4i*)(unsigned long long)l;
    #pragma unroll
    for (int i = 0; i < (UNITS + 255) / 256; ++i) {
        const int u = threadIdx.x + i * 256;
        if (UNITS % 256 == 0 || u < UNITS)
            __builtin_amdgcn_global_load_async_to_lds_b128(gp + u, lp + u, 0, 0);
    }
    wait_async0();
#else
    const bf16x8* gs = (const bf16x8*)g;
    bf16x8*       ls = (bf16x8*)l;
    #pragma unroll
    for (int i = 0; i < (UNITS + 255) / 256; ++i) {
        const int u = threadIdx.x + i * 256;
        if (UNITS % 256 == 0 || u < UNITS) ls[u] = gs[u];
    }
#endif
}

__device__ __forceinline__ int kchunk_map(int e, int laneHi) {
    return ((e < 8) ? e : (e + 8)) + (laneHi ? 8 : 0);
}

// bf16 A fragment: elements 0..7 at p[0..7], 8..15 at p[16..23].
__device__ __forceinline__ v16bf load_a_bf16(const __bf16* __restrict__ p) {
    bf16x8 lo = *(const bf16x8*)(p);
    bf16x8 hi = *(const bf16x8*)(p + 16);
    return __builtin_shufflevector(lo, hi, 0, 1, 2, 3, 4, 5, 6, 7,
                                           8, 9, 10, 11, 12, 13, 14, 15);
}

// ---------------------------------------------------------------------------
// Per-layer weight prep: write fragment-ordered bf16 panels to global.
// Dest offsets are exactly tid / tid+256 per K-chunk -> coalesced b16 stores.
// ---------------------------------------------------------------------------
template <int CIN, int COUT>
__global__ __launch_bounds__(256) void prep_conv_w_kernel(
    const float* __restrict__ w, __bf16* __restrict__ kp)
{
    constexpr int KTOT = 9 * CIN;
    const int ntile = blockIdx.x;
    const int tid = threadIdx.x, e = tid & 15, col = tid >> 4;
    const int off0 = (e < 8) ? e : e + 8;
    const float* up0 = w + (size_t)off0 * COUT + ntile * 16 + col;
    const float* up1 = up0 + 8 * COUT;
    __bf16* dp = kp + (size_t)ntile * (KTOT * 16);
    #pragma unroll
    for (int kc = 0; kc < KTOT / 32; ++kc) {
        dp[kc * 512 + tid]       = (__bf16)up0[kc * 32 * COUT];
        dp[kc * 512 + tid + 256] = (__bf16)up1[kc * 32 * COUT];
    }
}

template <int C>
__global__ __launch_bounds__(256) void prep_lstm_u_kernel(
    const float* __restrict__ u, __bf16* __restrict__ upp)
{
    constexpr int KTOT = 9 * C, C4 = 4 * C;
    const int ntile = blockIdx.x, g = blockIdx.y;
    const int tid = threadIdx.x, e = tid & 15, col = tid >> 4;
    const int off0 = (e < 8) ? e : e + 8;
    const float* up0 = u + (size_t)off0 * C4 + g * C + ntile * 16 + col;
    const float* up1 = up0 + 8 * C4;
    __bf16* dp = upp + ((size_t)ntile * 4 + g) * (KTOT * 16);
    #pragma unroll
    for (int kc = 0; kc < KTOT / 32; ++kc) {
        dp[kc * 512 + tid]       = (__bf16)up0[kc * 32 * C4];
        dp[kc * 512 + tid + 256] = (__bf16)up1[kc * 32 * C4];
    }
}

// ---------------------------------------------------------------------------
// Input conv for layers >= 1: bf16 halo-padded input, stride 2, branch-free.
// ---------------------------------------------------------------------------
template <int CIN, int COUT>
__global__ __launch_bounds__(256) void conv_in_bf16_kernel(
    const __bf16* __restrict__ x,     // padded bf16 input [NB][H+1][W+1][CIN]
    const __bf16* __restrict__ kp,    // pre-permuted panels [COUT/16][KTOT*16]
    const float* __restrict__ bias,   // [COUT]
    float* __restrict__ xg,           // [NB, Hp, Wp, COUT] f32
    int ldW, int plane, int Hp, int Wp, int lwp, int mtiles)
{
    constexpr int KTOT = 9 * CIN;
    constexpr int NCH  = CIN / 32;
    __shared__ __bf16 s_w[KTOT * 16];

    const int nBase = blockIdx.y * 16;
    const int img   = blockIdx.z;
    const int HpWp  = Hp * Wp;
    const __bf16* imgp = x  + (size_t)img * plane;
    float*        xgp  = xg + (size_t)img * ((size_t)HpWp * COUT);

    stage_lds<KTOT * 16 * 2>(kp + (size_t)blockIdx.y * (KTOT * 16), s_w);
    __syncthreads();

    const int lane   = threadIdx.x & 31;
    const int laneHi = lane >> 4;
    const int mtile  = blockIdx.x * 8 + (threadIdx.x >> 5);
    const int pixel  = mtile * 16 + (lane & 15);
    const bool pvalid = (mtile < mtiles) && (pixel < HpWp);
    const int pc = pvalid ? pixel : 0;
    const int oh = pc >> lwp, ow = pc & (Wp - 1);

    v8f acc = {};
    #pragma unroll
    for (int kh = 0; kh < 3; ++kh) {
        #pragma unroll
        for (int kw = 0; kw < 3; ++kw) {
            const unsigned poff =
                (unsigned)((oh * 2 + kh) * ldW + (ow * 2 + kw)) * CIN + laneHi * 8;
            #pragma unroll
            for (int cc = 0; cc < NCH; ++cc) {
                v16bf a = load_a_bf16(imgp + poff + cc * 32);
                const int kc = (kh * 3 + kw) * NCH + cc;
                v16bf bfr = *reinterpret_cast<const v16bf*>(&s_w[kc * 512 + lane * 16]);
                acc = __builtin_amdgcn_wmma_f32_16x16x32_bf16(
                          false, a, false, bfr, (short)0, acc, false, false);
            }
        }
    }

    const int n = lane & 15;
    const float bv = bias[nBase + n];
    const int m0 = laneHi ? 8 : 0;
    float* outp = xgp + (unsigned)(mtile * 16 + m0) * COUT + nBase + n;
    if (mtile * 16 + 16 <= HpWp) {
        #pragma unroll
        for (int r = 0; r < 8; ++r) outp[r * COUT] = acc[r] + bv;
    } else {
        #pragma unroll
        for (int r = 0; r < 8; ++r) {
            const int p = mtile * 16 + m0 + r;
            if (mtile < mtiles && p < HpWp) outp[r * COUT] = acc[r] + bv;
        }
    }
}

// Layer-0 input conv: f32 harness input, CIN=3, K=27 -> one padded chunk.
__global__ __launch_bounds__(256) void conv_in_c3_kernel(
    const float* __restrict__ x, const float* __restrict__ w,
    const float* __restrict__ bias, float* __restrict__ xg,
    int H, int W, int Hp, int Wp, int lwp, int mtiles)
{
    constexpr int COUT = 128;
    __shared__ __bf16 s_w[32 * 16];
    const int nBase = blockIdx.y * 16;
    const int img   = blockIdx.z;
    const int HpWp  = Hp * Wp;
    const float* imgp = x  + (size_t)img * ((size_t)H * W * 3);
    float*       xgp  = xg + (size_t)img * ((size_t)HpWp * COUT);

    for (int idx = threadIdx.x; idx < 512; idx += 256) {
        const int l2 = (idx >> 4) & 31;
        const int e  = idx & 15;
        const int k  = kchunk_map(e, l2 >> 4);
        const float v = (k < 27) ? w[k * COUT + nBase + (l2 & 15)] : 0.0f;
        s_w[idx] = (__bf16)v;
    }
    __syncthreads();

    const int lane   = threadIdx.x & 31;
    const int laneHi = lane >> 4;
    const int mtile  = blockIdx.x * 8 + (threadIdx.x >> 5);
    const int pixel  = mtile * 16 + (lane & 15);
    const bool pvalid = (mtile < mtiles) && (pixel < HpWp);
    const int pc = pvalid ? pixel : 0;
    const int oh = pc >> lwp, ow = pc & (Wp - 1);

    v16bf a;
    #pragma unroll
    for (int e = 0; e < 16; ++e) {
        const int k = kchunk_map(e, laneHi);
        float v = 0.0f;
        if (pvalid && k < 27) {
            const int tap = k / 3, cin = k - tap * 3;
            const int kh = tap / 3, kw = tap - (tap / 3) * 3;
            const int ih = oh * 2 + kh, iw = ow * 2 + kw;
            if (ih < H && iw < W)
                v = imgp[(unsigned)(ih * W + iw) * 3 + cin];
        }
        a[e] = (__bf16)v;
    }
    v16bf bfr = *reinterpret_cast<const v16bf*>(&s_w[lane * 16]);
    v8f acc = __builtin_amdgcn_wmma_f32_16x16x32_bf16(
                  false, a, false, bfr, (short)0, v8f{}, false, false);

    const int n = lane & 15;
    const float bv = bias[nBase + n];
    const int m0 = laneHi ? 8 : 0;
    float* outp = xgp + (unsigned)(mtile * 16 + m0) * COUT + nBase + n;
    if (mtile * 16 + 16 <= HpWp) {
        #pragma unroll
        for (int r = 0; r < 8; ++r) outp[r * COUT] = acc[r] + bv;
    } else {
        #pragma unroll
        for (int r = 0; r < 8; ++r) {
            const int p = mtile * 16 + m0 + r;
            if (mtile < mtiles && p < HpWp) outp[r * COUT] = acc[r] + bv;
        }
    }
}

// ---------------------------------------------------------------------------
// Fused LSTM step. h state bf16, halo padded. All four pre-permuted gate
// panels async-copied to dynamic LDS once; each A fragment feeds 4 WMMAs.
// ---------------------------------------------------------------------------
template <int C, bool LAST>
__global__ __launch_bounds__(256) void lstm_step_wmma_kernel(
    const float* __restrict__ xg,      // [B*T, Hp, Wp, 4C] f32 (bias folded)
    const __bf16* __restrict__ upp,    // [C/16][4][KTOT*16] permuted panels
    const __bf16* __restrict__ hprev,  // padded bf16, or nullptr (t=0)
    __bf16* __restrict__ hnew,         // padded bf16
    float* __restrict__ cst,           // [B, Hp, Wp, C] f32
    void* __restrict__ outv,           // LAST: f32 [B,T,HpWp,C]; else padded bf16
    const float* __restrict__ gamma, const float* __restrict__ beta,
    const float* __restrict__ mean,  const float* __restrict__ var,
    int t, int T, int Hp, int Wp, int lwp, int mtiles)
{
    constexpr int KTOT = 9 * C;
    constexpr int NCH  = C / 32;
    constexpr int C4   = 4 * C;
    extern __shared__ __bf16 s_w[];    // 4 * KTOT * 16 halves

    const int nb   = blockIdx.y * 16;
    const int b    = blockIdx.z;
    const int HpWp = Hp * Wp;
    const int W2   = Wp + 2;
    const unsigned hplane = (unsigned)(Hp + 2) * W2 * C;

    const int lane   = threadIdx.x & 31;
    const int laneHi = lane >> 4;
    const int mtile  = blockIdx.x * 8 + (threadIdx.x >> 5);
    const int pixel  = mtile * 16 + (lane & 15);
    const bool pvalid = (mtile < mtiles) && (pixel < HpWp);
    const int pc = pvalid ? pixel : 0;
    const int oh = pc >> lwp, ow = pc & (Wp - 1);

    v8f acc[4] = {};

    if (hprev) {
        stage_lds<4 * KTOT * 16 * 2>(upp + (size_t)blockIdx.y * (4 * KTOT * 16), s_w);
        __syncthreads();

        const __bf16* himg = hprev + (size_t)b * hplane;
        const int rowst = W2 * C;
        const unsigned abase = (unsigned)(oh * rowst) + (unsigned)ow * C + laneHi * 8;

        #pragma unroll
        for (int kh = 0; kh < 3; ++kh) {
            #pragma unroll
            for (int kw = 0; kw < 3; ++kw) {
                const __bf16* p0 = himg + abase + kh * rowst + kw * C;
                #pragma unroll
                for (int cc = 0; cc < NCH; ++cc) {
                    v16bf a = load_a_bf16(p0 + cc * 32);   // halo: no guard
                    const int kcb = ((kh * 3 + kw) * NCH + cc) * 512 + lane * 16;
                    #pragma unroll
                    for (int g = 0; g < 4; ++g) {
                        v16bf bfr = *reinterpret_cast<const v16bf*>(
                                        &s_w[g * (KTOT * 16) + kcb]);
                        acc[g] = __builtin_amdgcn_wmma_f32_16x16x32_bf16(
                                     false, a, false, bfr, (short)0, acc[g],
                                     false, false);
                    }
                }
            }
        }
    }

    // Fused epilogue: gates, cell update, BatchNorm.
    const int n  = lane & 15;
    const int ch = nb + n;
    const float sc = gamma[ch] * rsqrtf(var[ch] + 1e-3f);
    const float mn = mean[ch];
    const float bt = beta[ch];
    const int bt_idx = b * T + t;
    const float* xgb = xg + (size_t)bt_idx * HpWp * C4;
    float*  cstb = cst  + (size_t)b * HpWp * C;
    __bf16* hnb  = hnew + (size_t)b * hplane;
    float*  outf = LAST ? ((float*)outv + (size_t)bt_idx * HpWp * C) : nullptr;
    __bf16* outh = LAST ? nullptr
                        : ((__bf16*)outv + (size_t)bt_idx * (Hp + 1) * (Wp + 1) * C);
    const int m0 = laneHi ? 8 : 0;

#define LSTM_EPI_BODY(p_)                                                     \
    {                                                                         \
        const unsigned xi = (unsigned)(p_) * C4;                              \
        float zi = acc[0][r] + xgb[xi + 0 * C + ch];                          \
        float zf = acc[1][r] + xgb[xi + 1 * C + ch];                          \
        float zc = acc[2][r] + xgb[xi + 2 * C + ch];                          \
        float zo = acc[3][r] + xgb[xi + 3 * C + ch];                          \
        float ig = hsig(zi), fg = hsig(zf), og = hsig(zo);                    \
        const unsigned ci = (unsigned)(p_) * C + ch;                          \
        float co = cstb[ci];                                                  \
        float cn = fg * co + ig * fast_tanh(zc);                              \
        float hn = og * fast_tanh(cn);                                        \
        cstb[ci] = cn;                                                        \
        const int oh2 = (p_) >> lwp, ow2 = (p_) & (Wp - 1);                   \
        hnb[(unsigned)((oh2 + 1) * W2 + (ow2 + 1)) * C + ch] = (__bf16)hn;    \
        const float bn = (hn - mn) * sc + bt;                                 \
        if (LAST) outf[ci] = bn;                                              \
        else outh[(unsigned)(oh2 * (Wp + 1) + ow2) * C + ch] = (__bf16)bn;    \
    }

    if (mtile * 16 + 16 <= HpWp) {                 // wave-uniform fast path
        #pragma unroll
        for (int r = 0; r < 8; ++r) {
            const int p = mtile * 16 + m0 + r;
            LSTM_EPI_BODY(p)
        }
    } else {
        #pragma unroll
        for (int r = 0; r < 8; ++r) {
            const int p = mtile * 16 + m0 + r;
            if (mtile < mtiles && p < HpWp) LSTM_EPI_BODY(p)
        }
    }
#undef LSTM_EPI_BODY
}

__global__ void zero_u32_kernel(unsigned* __restrict__ p, long long n) {
    long long i = (long long)blockIdx.x * blockDim.x + threadIdx.x;
    if (i < n) p[i] = 0u;
}

// ---------------------------------------------------------------------------
// Host orchestration (graph-capture safe).
// ---------------------------------------------------------------------------
extern "C" void kernel_launch(void* const* d_in, const int* in_sizes, int n_in,
                              void* d_out, int out_size, void* d_ws, size_t ws_size,
                              hipStream_t stream) {
    (void)in_sizes; (void)n_in; (void)out_size; (void)ws_size;

    const int B = 4, T = 8, NB = B * T;
    const int NLAYERS = 7;
    const int CH[NLAYERS] = {32, 64, 64, 128, 128, 128, 128};

    const float* x = (const float*)d_in[0];
    const float* kern[NLAYERS], *recur[NLAYERS], *bias[NLAYERS];
    const float* gam[NLAYERS], *bet[NLAYERS], *mea[NLAYERS], *va[NLAYERS];
    for (int i = 0; i < NLAYERS; ++i) {
        kern[i]  = (const float*)d_in[1 + i];
        recur[i] = (const float*)d_in[8 + i];
        bias[i]  = (const float*)d_in[15 + i];
        gam[i]   = (const float*)d_in[22 + i];
        bet[i]   = (const float*)d_in[29 + i];
        mea[i]   = (const float*)d_in[36 + i];
        va[i]    = (const float*)d_in[43 + i];
    }

    // Workspace: xg f32 | pingA bf16 | pingB bf16 | h0 | h1 | c f32 | kp | up
    char* base = (char*)d_ws;
    const size_t XG_MAX   = (size_t)NB * 128 * 128 * 128;   // f32 elems
    const size_t PING_MAX = (size_t)NB * 129 * 129 * 32;    // bf16 elems
    const size_t HPAD_MAX = (size_t)B  * 130 * 130 * 32;    // bf16 elems
    const size_t CST_MAX  = (size_t)B  * 128 * 128 * 32;    // f32 elems
    const size_t PREP_MAX = (size_t)32 * 1152 * 16;         // bf16 elems (589824)
    float*  xg    = (float*)base;
    __bf16* pingA = (__bf16*)(base + XG_MAX * 4);
    __bf16* pingB = pingA + PING_MAX;
    __bf16* h0    = pingB + PING_MAX;
    __bf16* h1    = h0 + HPAD_MAX;
    float*  cst   = (float*)(h1 + HPAD_MAX);
    __bf16* kprep = (__bf16*)(cst + CST_MAX);
    __bf16* uprep = kprep + PREP_MAX;

    int Hc = 256, Wc = 256, Cin = 3;
    const __bf16* prev_out = nullptr;

    for (int l = 0; l < NLAYERS; ++l) {
        const int C = CH[l];
        const int C4 = 4 * C;
        const int Hp = Hc / 2, Wp = Wc / 2;
        const int HpWp = Hp * Wp;
        const int lwp = 31 - __builtin_clz((unsigned)Wp);
        const int mtiles = (HpWp + 15) / 16;
        const int mblocks = (mtiles + 7) / 8;
        const bool last = (l == NLAYERS - 1);

        __bf16* outb = (l & 1) ? pingB : pingA;   // padded [NB][Hp+1][Wp+1][C]

        // 0) Per-layer weight prep (fragment-ordered bf16 panels in global).
        if (l > 0) {
            dim3 gp(C4 / 16);
            if (Cin == 32)      prep_conv_w_kernel<32, 256><<<gp, dim3(256), 0, stream>>>(kern[l], kprep);
            else if (Cin == 64 && C4 == 256)
                                prep_conv_w_kernel<64, 256><<<gp, dim3(256), 0, stream>>>(kern[l], kprep);
            else if (Cin == 64) prep_conv_w_kernel<64, 512><<<gp, dim3(256), 0, stream>>>(kern[l], kprep);
            else                prep_conv_w_kernel<128, 512><<<gp, dim3(256), 0, stream>>>(kern[l], kprep);
        }
        {
            dim3 gu(C / 16, 4);
            if (C == 32)      prep_lstm_u_kernel<32><<<gu, dim3(256), 0, stream>>>(recur[l], uprep);
            else if (C == 64) prep_lstm_u_kernel<64><<<gu, dim3(256), 0, stream>>>(recur[l], uprep);
            else              prep_lstm_u_kernel<128><<<gu, dim3(256), 0, stream>>>(recur[l], uprep);
        }

        // 1) Batched input conv over all B*T images -> xg (f32, bias folded).
        dim3 gci(mblocks, C4 / 16, NB);
        if (l == 0) {
            conv_in_c3_kernel<<<gci, dim3(256), 0, stream>>>(
                x, kern[l], bias[l], xg, Hc, Wc, Hp, Wp, lwp, mtiles);
        } else {
            const int ldW = Wc + 1;
            const int plane = (Hc + 1) * (Wc + 1) * Cin;
            if (Cin == 32) {
                conv_in_bf16_kernel<32, 256><<<gci, dim3(256), 0, stream>>>(
                    prev_out, kprep, bias[l], xg, ldW, plane, Hp, Wp, lwp, mtiles);
            } else if (Cin == 64 && C4 == 256) {
                conv_in_bf16_kernel<64, 256><<<gci, dim3(256), 0, stream>>>(
                    prev_out, kprep, bias[l], xg, ldW, plane, Hp, Wp, lwp, mtiles);
            } else if (Cin == 64) {
                conv_in_bf16_kernel<64, 512><<<gci, dim3(256), 0, stream>>>(
                    prev_out, kprep, bias[l], xg, ldW, plane, Hp, Wp, lwp, mtiles);
            } else {
                conv_in_bf16_kernel<128, 512><<<gci, dim3(256), 0, stream>>>(
                    prev_out, kprep, bias[l], xg, ldW, plane, Hp, Wp, lwp, mtiles);
            }
        }

        // 2) Zero h0|h1 (bf16), cell state (f32), padded layer-output ring.
        const size_t hpad = (size_t)B * (Hp + 2) * (Wp + 2) * C;   // bf16 elems
        const long long hw = (long long)hpad;                      // u32 words
        zero_u32_kernel<<<dim3((unsigned)((hw + 255) / 256)), dim3(256), 0, stream>>>(
            (unsigned*)h0, hw);
        const long long cw = (long long)B * HpWp * C;
        zero_u32_kernel<<<dim3((unsigned)((cw + 255) / 256)), dim3(256), 0, stream>>>(
            (unsigned*)cst, cw);
        if (!last) {
            const long long ow2 = (long long)NB * (Hp + 1) * (Wp + 1) * C / 2;
            zero_u32_kernel<<<dim3((unsigned)((ow2 + 255) / 256)), dim3(256), 0, stream>>>(
                (unsigned*)outb, ow2);
        }

        // 3) Sequential recurrence, fully fused per step.
        const size_t lds = (size_t)4 * 9 * C * 16 * 2;   // 4 gate panels, bf16
        __bf16* hp = nullptr;
        __bf16* hn = h0;
        for (int t = 0; t < T; ++t) {
            dim3 gst(mblocks, C / 16, B);
            void* outp = last ? (void*)d_out : (void*)outb;
            if (C == 32) {
                lstm_step_wmma_kernel<32, false><<<gst, dim3(256), lds, stream>>>(
                    xg, uprep, hp, hn, cst, outp,
                    gam[l], bet[l], mea[l], va[l], t, T, Hp, Wp, lwp, mtiles);
            } else if (C == 64) {
                lstm_step_wmma_kernel<64, false><<<gst, dim3(256), lds, stream>>>(
                    xg, uprep, hp, hn, cst, outp,
                    gam[l], bet[l], mea[l], va[l], t, T, Hp, Wp, lwp, mtiles);
            } else if (!last) {
                lstm_step_wmma_kernel<128, false><<<gst, dim3(256), lds, stream>>>(
                    xg, uprep, hp, hn, cst, outp,
                    gam[l], bet[l], mea[l], va[l], t, T, Hp, Wp, lwp, mtiles);
            } else {
                lstm_step_wmma_kernel<128, true><<<gst, dim3(256), lds, stream>>>(
                    xg, uprep, hp, hn, cst, outp,
                    gam[l], bet[l], mea[l], va[l], t, T, Hp, Wp, lwp, mtiles);
            }
            hp = hn;
            hn = (hp == h0) ? h1 : h0;
        }

        prev_out = outb;
        Hc = Hp; Wc = Wp; Cin = C;
    }
}
